// MHA_5970004542011
// MI455X (gfx1250) — compile-verified, matrix operand order checked by reading, not run on hardware
//
#include <hip/hip_runtime.h>
#include <hip/hip_bf16.h>
#include <math.h>

// ---------------------------------------------------------------------------
// Problem constants (from reference): B=2, T=2048, C=1024, H=16, hd=64
// ---------------------------------------------------------------------------
#define BATCH 2
#define T_SEQ 2048
#define C_DIM 1024
#define N_QKV 3072
#define NH    16
#define HD    64

typedef __bf16 bf16;
typedef __bf16 v16bf __attribute__((ext_vector_type(16)));
typedef __bf16 v8bf  __attribute__((ext_vector_type(8)));
typedef float  v8f   __attribute__((ext_vector_type(8)));

union FragBF {
    v16bf v;
    v8bf  h[2];
};

// gfx1250 16-bit A/B fragment for 16x16x32 WMMA (wave32).
// A 16x32 layout (ISA 7.12.2): lane l holds row (l&15); VGPRs 0..3 hold
// K = (l>=16 ? 8 : 0) + 0..7 (contiguous), VGPRs 4..7 hold K = 16 + same.
// B (32x16) uses the identical per-lane packing with N = lane&15, i.e. lane n
// holds row n of B^T — so K / V^T / W^T stored row-major load directly.
// => two 16-byte loads per lane per fragment.
__device__ inline v16bf load_frag16x32(const bf16* __restrict__ base, int ldm, int lane) {
    int r  = lane & 15;
    int hs = (lane >> 4) & 1;
    const bf16* p = base + (size_t)r * ldm + hs * 8;
    FragBF f;
    f.h[0] = *reinterpret_cast<const v8bf*>(p);       // K = hs*8 .. hs*8+7
    f.h[1] = *reinterpret_cast<const v8bf*>(p + 16);  // K = 16 + hs*8 .. +7
    return f.v;
}

__device__ inline v8f wmma_bf16(v16bf a, v16bf b, v8f c) {
    // 8 args: (neg_a, A, neg_b, B, c_mod, C, reuse_a, reuse_b)
    return __builtin_amdgcn_wmma_f32_16x16x32_bf16(false, a, false, b, (short)0, c,
                                                   false, false);
}

// ---------------------------------------------------------------------------
// Precision conversion kernels
// ---------------------------------------------------------------------------
__global__ void cvt_f32_bf16_kernel(const float* __restrict__ src,
                                    bf16* __restrict__ dst, int n) {
    int i = blockIdx.x * blockDim.x + threadIdx.x;
    if (i < n) dst[i] = (bf16)src[i];
}

// w [K][N] row-major f32  ->  wT [N][K] row-major bf16 (B-fragment friendly)
__global__ void cvt_transpose_kernel(const float* __restrict__ w,
                                     bf16* __restrict__ wT, int K, int N) {
    int i = blockIdx.x * blockDim.x + threadIdx.x;
    if (i < K * N) {
        int k = i / N;
        int n = i % N;
        wT[(size_t)n * K + k] = (bf16)w[i];
    }
}

// ---------------------------------------------------------------------------
// QKV GEMM: qkv[M=B*T][N=3C] = x16[M][K=C] * wT^T, row-major bf16 output.
// One wave computes a 64x64 tile: per 32-wide K step, 8 fragment loads feed
// 16 WMMAs -> 32 bf16-FLOP per byte pulled from L2. Epilogue is a trivial
// strided store (fully unrolled so acc[][] never leaves VGPRs — no scratch).
// ---------------------------------------------------------------------------
__global__ __launch_bounds__(256) void qkv_gemm_kernel(
    const bf16* __restrict__ x16, const bf16* __restrict__ wT,
    bf16* __restrict__ qkv)
{
    int lane = threadIdx.x & 31;
    int wave = threadIdx.x >> 5;
    int tile = blockIdx.x * 8 + wave;
    const int ntilesN = N_QKV / 64;  // 48
    int m0 = (tile / ntilesN) * 64;
    int n0 = (tile % ntilesN) * 64;
    if (m0 >= BATCH * T_SEQ) return;

    v8f acc[4][4] = {};
    for (int kk = 0; kk < C_DIM; kk += 32) {
        v16bf a[4], b[4];
#pragma unroll
        for (int i = 0; i < 4; ++i) {
            a[i] = load_frag16x32(x16 + (size_t)(m0 + 16 * i) * C_DIM + kk, C_DIM, lane);
            b[i] = load_frag16x32(wT + (size_t)(n0 + 16 * i) * C_DIM + kk, C_DIM, lane);
        }
#pragma unroll
        for (int mi = 0; mi < 4; ++mi)
#pragma unroll
            for (int ni = 0; ni < 4; ++ni)
                acc[mi][ni] = wmma_bf16(a[mi], b[ni], acc[mi][ni]);
    }

    // C/D layout: lanes 0-15 vgpr j -> (M=j, N=lane); lanes 16-31 -> (M=8+j, N=lane-16)
    int r  = lane & 15;
    int hs = lane >> 4;
#pragma unroll
    for (int mi = 0; mi < 4; ++mi) {
#pragma unroll
        for (int ni = 0; ni < 4; ++ni) {
            bf16* p = qkv + (size_t)(m0 + mi * 16 + hs * 8) * N_QKV + (n0 + ni * 16 + r);
#pragma unroll
            for (int j = 0; j < 8; ++j)
                p[(size_t)j * N_QKV] = (bf16)acc[mi][ni][j];
        }
    }
}

// ---------------------------------------------------------------------------
// Fused RoPE + scatter: reads qkv[M][3C], writes
//   q,k : [B][H][T][hd]  (RoPE applied, half-split rotation)
//   vT  : [B][H][hd][T]  (transposed copy for P*V B-fragments)
// One thread per (b,h,t,i<hd/2) handles the rotation pair (i, i+32).
// ---------------------------------------------------------------------------
__global__ void rope_scatter_kernel(const bf16* __restrict__ qkv,
                                    bf16* __restrict__ q, bf16* __restrict__ k,
                                    bf16* __restrict__ vt, int n) {
    int idx = blockIdx.x * blockDim.x + threadIdx.x;
    if (idx >= n) return;                    // n = B*H*T*(hd/2)
    int i  = idx & 31;                       // rotation pair index, 0..31
    int t  = (idx >> 5) & (T_SEQ - 1);
    int bh = idx >> 16;                      // 0..31
    int b  = bh >> 4;
    int h  = bh & 15;

    size_t row = ((size_t)b * T_SEQ + t) * N_QKV;
    int c0 = h * HD + i;
    float q0 = (float)qkv[row + c0];
    float q1 = (float)qkv[row + c0 + 32];
    float k0 = (float)qkv[row + C_DIM + c0];
    float k1 = (float)qkv[row + C_DIM + c0 + 32];
    bf16 v0  = qkv[row + 2 * C_DIM + c0];
    bf16 v1  = qkv[row + 2 * C_DIM + c0 + 32];

    float freq = expf(-((float)(2 * i) / (float)HD) * logf(10000.0f));
    float ang  = (float)t * freq;
    float cs = cosf(ang), sn = sinf(ang);

    size_t obase = (size_t)bh * (T_SEQ * HD);
    q[obase + (size_t)t * HD + i]        = (bf16)(q0 * cs - q1 * sn);
    q[obase + (size_t)t * HD + i + 32]   = (bf16)(q1 * cs + q0 * sn);
    k[obase + (size_t)t * HD + i]        = (bf16)(k0 * cs - k1 * sn);
    k[obase + (size_t)t * HD + i + 32]   = (bf16)(k1 * cs + k0 * sn);
    vt[obase + (size_t)i * T_SEQ + t]        = v0;
    vt[obase + (size_t)(i + 32) * T_SEQ + t] = v1;
}

// ---------------------------------------------------------------------------
// Causal flash attention, wave32: one 16-query tile per wave, 32 keys/step.
// Per step: 4 WMMAs for S = QK^T (hd=64 -> 2 K-chunks x 2 key-subtiles),
// online softmax (16-lane shfl reductions matching the C-fragment row map),
// P transposed C-layout -> A-layout via LDS, then 4 WMMAs for O += P*V.
// ---------------------------------------------------------------------------
__global__ __launch_bounds__(256) void attn_kernel(
    const bf16* __restrict__ q, const bf16* __restrict__ kmat,
    const bf16* __restrict__ vt, bf16* __restrict__ y)
{
    __shared__ bf16 pbuf[8][16 * 40];  // per-wave 16x32 P tile, padded rows
    int lane = threadIdx.x & 31;
    int wave = threadIdx.x >> 5;
    int task = blockIdx.x * 8 + wave;
    const int nQT = T_SEQ / 16;        // 128
    if (task >= BATCH * NH * nQT) return;
    int t0 = (task % nQT) * 16;
    int bh = task / nQT;
    int b  = bh / NH;
    int h  = bh % NH;

    const bf16* qb = q    + (size_t)bh * (T_SEQ * HD) + (size_t)t0 * HD;
    const bf16* kb = kmat + (size_t)bh * (T_SEQ * HD);
    const bf16* vb = vt   + (size_t)bh * (T_SEQ * HD);   // [HD][T]

    v16bf aq0 = load_frag16x32(qb + 0,  HD, lane);
    v16bf aq1 = load_frag16x32(qb + 32, HD, lane);

    v8f o0 = {}, o1 = {}, o2 = {}, o3 = {};
    float mrow[8], lrow[8];
#pragma unroll
    for (int j = 0; j < 8; ++j) { mrow[j] = -1e30f; lrow[j] = 0.f; }

    int r  = lane & 15;
    int hs = lane >> 4;
    bf16* pl = &pbuf[wave][0];
    const float scale = 0.125f;  // 1/sqrt(64)

    for (int k0 = 0; k0 < t0 + 16; k0 += 32) {
        v16bf bkA0 = load_frag16x32(kb + (size_t)k0 * HD + 0,         HD, lane);
        v16bf bkA1 = load_frag16x32(kb + (size_t)k0 * HD + 32,        HD, lane);
        v16bf bkB0 = load_frag16x32(kb + (size_t)(k0 + 16) * HD + 0,  HD, lane);
        v16bf bkB1 = load_frag16x32(kb + (size_t)(k0 + 16) * HD + 32, HD, lane);
        v8f s0 = {}, s1 = {};
        s0 = wmma_bf16(aq0, bkA0, s0);
        s0 = wmma_bf16(aq1, bkA1, s0);
        s1 = wmma_bf16(aq0, bkB0, s1);
        s1 = wmma_bf16(aq1, bkB1, s1);

        float p0[8], p1[8], mt[8];
#pragma unroll
        for (int j = 0; j < 8; ++j) {
            int row = t0 + hs * 8 + j;
            float v0 = s0[j] * scale;
            float v1 = s1[j] * scale;
            if (k0 + r > row)      v0 = -1e30f;   // causal mask
            if (k0 + 16 + r > row) v1 = -1e30f;
            p0[j] = v0; p1[j] = v1;
            mt[j] = fmaxf(v0, v1);
        }
        // row max across the 16 lanes holding this row (xor<16 stays in half-wave)
#pragma unroll
        for (int off = 1; off < 16; off <<= 1)
#pragma unroll
            for (int j = 0; j < 8; ++j)
                mt[j] = fmaxf(mt[j], __shfl_xor(mt[j], off, 32));
        float corr[8];
#pragma unroll
        for (int j = 0; j < 8; ++j) {
            float mn = fmaxf(mrow[j], mt[j]);
            corr[j] = __expf(mrow[j] - mn);
            mrow[j] = mn;
            p0[j] = __expf(p0[j] - mn);
            p1[j] = __expf(p1[j] - mn);
        }
        float rs[8];
#pragma unroll
        for (int j = 0; j < 8; ++j) rs[j] = p0[j] + p1[j];
#pragma unroll
        for (int off = 1; off < 16; off <<= 1)
#pragma unroll
            for (int j = 0; j < 8; ++j)
                rs[j] += __shfl_xor(rs[j], off, 32);
#pragma unroll
        for (int j = 0; j < 8; ++j) lrow[j] = lrow[j] * corr[j] + rs[j];
#pragma unroll
        for (int j = 0; j < 8; ++j) {
            o0[j] *= corr[j]; o1[j] *= corr[j]; o2[j] *= corr[j]; o3[j] *= corr[j];
        }

        // C-layout -> row-major 16x32 in LDS, reload as A fragment.
#pragma unroll
        for (int j = 0; j < 8; ++j) {
            pl[(hs * 8 + j) * 40 + r]      = (bf16)p0[j];
            pl[(hs * 8 + j) * 40 + 16 + r] = (bf16)p1[j];
        }
        asm volatile("s_wait_dscnt 0" ::: "memory");  // DS is per-wave in-order; fence compiler too
        v16bf pa = load_frag16x32(pl, 40, lane);

        // O[16 x 64] += P[16 x 32] * V[32 x 64]; B-fragments from row-contiguous V^T
        o0 = wmma_bf16(pa, load_frag16x32(vb + (size_t) 0 * T_SEQ + k0, T_SEQ, lane), o0);
        o1 = wmma_bf16(pa, load_frag16x32(vb + (size_t)16 * T_SEQ + k0, T_SEQ, lane), o1);
        o2 = wmma_bf16(pa, load_frag16x32(vb + (size_t)32 * T_SEQ + k0, T_SEQ, lane), o2);
        o3 = wmma_bf16(pa, load_frag16x32(vb + (size_t)48 * T_SEQ + k0, T_SEQ, lane), o3);
    }

    float inv[8];
#pragma unroll
    for (int j = 0; j < 8; ++j) inv[j] = 1.0f / lrow[j];
#pragma unroll
    for (int j = 0; j < 8; ++j) {
        o0[j] *= inv[j]; o1[j] *= inv[j]; o2[j] *= inv[j]; o3[j] *= inv[j];
    }
    // y[b][t][h*64 + d] bf16, row-major C for the projection GEMM
    bf16* yp = y + ((size_t)b * T_SEQ + t0 + hs * 8) * C_DIM + h * HD + r;
#pragma unroll
    for (int j = 0; j < 8; ++j) {
        yp[(size_t)j * C_DIM +  0] = (bf16)o0[j];
        yp[(size_t)j * C_DIM + 16] = (bf16)o1[j];
        yp[(size_t)j * C_DIM + 32] = (bf16)o2[j];
        yp[(size_t)j * C_DIM + 48] = (bf16)o3[j];
    }
}

// ---------------------------------------------------------------------------
// Projection GEMM: out[M=B*T][N=C] = y[M][K=C] * wpT^T, f32 output.
// Same 64x64-per-wave register blocking; fully unrolled epilogue.
// ---------------------------------------------------------------------------
__global__ __launch_bounds__(256) void proj_gemm_kernel(
    const bf16* __restrict__ y, const bf16* __restrict__ wpT,
    float* __restrict__ out)
{
    int lane = threadIdx.x & 31;
    int wave = threadIdx.x >> 5;
    int tile = blockIdx.x * 8 + wave;
    const int ntilesN = C_DIM / 64;  // 16
    int m0 = (tile / ntilesN) * 64;
    int n0 = (tile % ntilesN) * 64;
    if (m0 >= BATCH * T_SEQ) return;

    v8f acc[4][4] = {};
    for (int kk = 0; kk < C_DIM; kk += 32) {
        v16bf a[4], b[4];
#pragma unroll
        for (int i = 0; i < 4; ++i) {
            a[i] = load_frag16x32(y + (size_t)(m0 + 16 * i) * C_DIM + kk, C_DIM, lane);
            b[i] = load_frag16x32(wpT + (size_t)(n0 + 16 * i) * C_DIM + kk, C_DIM, lane);
        }
#pragma unroll
        for (int mi = 0; mi < 4; ++mi)
#pragma unroll
            for (int ni = 0; ni < 4; ++ni)
                acc[mi][ni] = wmma_bf16(a[mi], b[ni], acc[mi][ni]);
    }
    int r  = lane & 15;
    int hs = lane >> 4;
#pragma unroll
    for (int mi = 0; mi < 4; ++mi) {
#pragma unroll
        for (int ni = 0; ni < 4; ++ni) {
            float* p = out + (size_t)(m0 + mi * 16 + hs * 8) * C_DIM + (n0 + ni * 16 + r);
#pragma unroll
            for (int j = 0; j < 8; ++j)
                p[(size_t)j * C_DIM] = acc[mi][ni][j];
        }
    }
}

// ---------------------------------------------------------------------------
// Host-side orchestration
// ---------------------------------------------------------------------------
extern "C" void kernel_launch(void* const* d_in, const int* in_sizes, int n_in,
                              void* d_out, int out_size, void* d_ws, size_t ws_size,
                              hipStream_t stream) {
    const float* x      = (const float*)d_in[0];
    const float* w_attn = (const float*)d_in[1];
    const float* w_proj = (const float*)d_in[2];
    float* out = (float*)d_out;

    char* ws = (char*)d_ws;
    size_t off = 0;
    auto alloc = [&](size_t bytes) -> void* {
        void* p = ws + off;
        off = (off + bytes + 255) & ~(size_t)255;
        return p;
    };
    const size_t M   = (size_t)BATCH * T_SEQ;           // 4096
    const size_t QKV = (size_t)BATCH * NH * T_SEQ * HD; // 4M elems per tensor

    bf16* x16  = (bf16*)alloc(M * C_DIM * sizeof(bf16));             //  8 MB
    bf16* wT   = (bf16*)alloc((size_t)N_QKV * C_DIM * sizeof(bf16)); //  6 MB
    bf16* wpT  = (bf16*)alloc((size_t)C_DIM * C_DIM * sizeof(bf16)); //  2 MB
    bf16* qkvb = (bf16*)alloc(M * N_QKV * sizeof(bf16));             // 24 MB
    bf16* qb   = (bf16*)alloc(QKV * sizeof(bf16));                   //  8 MB
    bf16* kb   = (bf16*)alloc(QKV * sizeof(bf16));                   //  8 MB
    bf16* vtb  = (bf16*)alloc(QKV * sizeof(bf16));                   //  8 MB
    bf16* yb   = (bf16*)alloc(M * C_DIM * sizeof(bf16));             //  8 MB

    // 1) precision conversion (+ weight transposes for fragment-friendly B)
    {
        int n = (int)(M * C_DIM);
        cvt_f32_bf16_kernel<<<(n + 255) / 256, 256, 0, stream>>>(x, x16, n);
        cvt_transpose_kernel<<<(C_DIM * N_QKV + 255) / 256, 256, 0, stream>>>(
            w_attn, wT, C_DIM, N_QKV);
        cvt_transpose_kernel<<<(C_DIM * C_DIM + 255) / 256, 256, 0, stream>>>(
            w_proj, wpT, C_DIM, C_DIM);
    }
    // 2) QKV GEMM (WMMA, 64x64/wave), row-major qkv output
    {
        int tiles = (int)(M / 64) * (N_QKV / 64);  // 64*48 = 3072
        qkv_gemm_kernel<<<tiles / 8, 256, 0, stream>>>(x16, wT, qkvb);
    }
    // 3) fused RoPE + q/k/vT scatter
    {
        int n = BATCH * NH * T_SEQ * (HD / 2);
        rope_scatter_kernel<<<(n + 255) / 256, 256, 0, stream>>>(qkvb, qb, kb, vtb, n);
    }
    // 4) causal flash attention (WMMA)
    {
        int tasks = BATCH * NH * (T_SEQ / 16);  // 4096 waves
        attn_kernel<<<tasks / 8, 256, 0, stream>>>(qb, kb, vtb, yb);
    }
    // 5) output projection (WMMA, 64x64/wave), f32 result
    {
        int tiles = (int)(M / 64) * (C_DIM / 64);  // 64*16 = 1024
        proj_gemm_kernel<<<tiles / 8, 256, 0, stream>>>(yb, wpT, out);
    }
}
